// LLamaSmoothquantAttention_51659866636888
// MI455X (gfx1250) — 8-bit matrix operand order repaired and checked by reading, compiled, not run on hardware
//
#include <hip/hip_runtime.h>
#include <hip/hip_bf16.h>
#include <math.h>

#define Sd   2048
#define Hd   2048
#define NHd  16
#define HDd  128

#define ROT_SCALE   0.01f
#define Q_OUT_SCALE 0.01f
#define K_OUT_SCALE 0.01f
#define LIN_ALPHA   (1.0f / 4096.0f)
#define LIN_BETA    1.0f
#define QK_ALPHA    (0.0001f / 11.313708498984761f)
#define PV_ALPHA    (1.0f / 127.0f)
#define OUT_ALPHA   (1.0f / 8192.0f)

typedef __attribute__((ext_vector_type(8))) int v8i;
typedef __attribute__((ext_vector_type(4))) int v4i;
typedef __attribute__((ext_vector_type(4))) unsigned int v4u;

#if defined(__has_builtin)
#if __has_builtin(__builtin_amdgcn_tensor_load_to_lds) && \
    __has_builtin(__builtin_amdgcn_s_wait_tensorcnt)
#define HAVE_TDM 1
#endif
#endif
#ifndef HAVE_TDM
#define HAVE_TDM 0
#endif

__device__ __forceinline__ int sat8(float f) {
  int r = (int)rintf(f);
  r = r > 127 ? 127 : r;
  r = r < -128 ? -128 : r;
  return r;
}

__device__ __forceinline__ v8i zero_v8i() {
  v8i z = {0, 0, 0, 0, 0, 0, 0, 0};
  return z;
}

__device__ __forceinline__ v8i wmma_iu8(v8i a, v8i b, v8i c) {
  return __builtin_amdgcn_wmma_i32_16x16x64_iu8(true, a, true, b, c, false, false);
}

__device__ __forceinline__ v8i load_frag_a(const signed char* base, int ld,
                                           int row0, int k0) {
  int lane = threadIdx.x & 31;
  const signed char* p =
      base + (size_t)(row0 + (lane & 15)) * (size_t)ld + k0 + ((lane >> 4) << 4);
  int4 c0 = *(const int4*)(p);
  int4 c1 = *(const int4*)(p + 32);
  v8i f = {c0.x, c0.y, c0.z, c0.w, c1.x, c1.y, c1.z, c1.w};
  return f;
}

__device__ __forceinline__ v8i load_frag_b(const signed char* bt, int ld,
                                           int col0, int k0) {
  int lane = threadIdx.x & 31;
  const signed char* p =
      bt + (size_t)(col0 + (lane & 15)) * (size_t)ld + k0 + ((lane >> 4) << 4);
  int4 c0 = *(const int4*)(p);
  int4 c1 = *(const int4*)(p + 32);
  v8i f = {c0.x, c0.y, c0.z, c0.w, c1.x, c1.y, c1.z, c1.w};
  return f;
}

#if HAVE_TDM
__device__ __forceinline__ void tdm_load_2d(unsigned lds_off,
                                            const signed char* gptr,
                                            unsigned tile_k, unsigned tile_rows,
                                            unsigned long long row_stride) {
  unsigned long long ga = (unsigned long long)(size_t)gptr;
  v4u g0;
  g0[0] = 1u;
  g0[1] = lds_off;
  g0[2] = (unsigned)(ga & 0xffffffffu);
  g0[3] = (unsigned)((ga >> 32) & 0x01ffffffu) | 0x80000000u;
  const unsigned td0 = 0x7fffffffu;
  const unsigned td1 = 0x7fffffffu;
  v8i g1;
  g1[0] = 0;
  g1[1] = (int)((td0 & 0xffffu) << 16);
  g1[2] = (int)(((td0 >> 16) & 0xffffu) | ((td1 & 0xffffu) << 16));
  g1[3] = (int)(((td1 >> 16) & 0xffffu) | ((tile_k & 0xffffu) << 16));
  g1[4] = (int)(tile_rows & 0xffffu);
  g1[5] = (int)(row_stride & 0xffffffffu);
  g1[6] = (int)((row_stride >> 32) & 0xffffu);
  g1[7] = 0;
  v4i z4 = {0, 0, 0, 0};
  v8i z8 = zero_v8i();
  __builtin_amdgcn_tensor_load_to_lds(g0, g1, z4, z4, z8, 0);
}
#endif

__device__ __forceinline__ void coop_copy_tile(signed char* dst,
                                               const signed char* src, int rows,
                                               int ldsrc) {
  int nchunks = rows * 4;
  for (int t = threadIdx.x; t < nchunks; t += 256) {
    int r = t >> 2, c = (t & 3) << 4;
    *(int4*)(dst + r * 64 + c) = *(const int4*)(src + (size_t)r * ldsrc + c);
  }
}

__global__ void pack_i8_kernel(const int* __restrict__ src,
                               signed char* __restrict__ dst, int n) {
  int i = blockIdx.x * blockDim.x + threadIdx.x;
  if (i < n) dst[i] = (signed char)src[i];
}

#define GEMM_BODY(X, W, K)                                                     \
  __shared__ __align__(16) signed char sA[2][128 * 64];                        \
  __shared__ __align__(16) signed char sB[2][256 * 64];                        \
  int tid = threadIdx.x, wid = tid >> 5, lane = tid & 31;                      \
  int m = lane & 15, half8 = (lane >> 4) << 3;                                 \
  int rowB = blockIdx.x * 128, colB = blockIdx.y * 256;                        \
  int rw = (wid >> 2) * 64, cw = (wid & 3) * 64;        \
  const int NK = (K) / 64;                                                     \
  v8i acc[4][4];                                                               \
  _Pragma("unroll") for (int i = 0; i < 4; ++i)                                \
      _Pragma("unroll") for (int j = 0; j < 4; ++j) acc[i][j] = zero_v8i();

__global__ void __launch_bounds__(256)
gemm_s8_req_kernel(const signed char* __restrict__ X,
                   const signed char* __restrict__ W,
                   const signed char* __restrict__ bias,
                   signed char* __restrict__ out, int K, int N) {
  GEMM_BODY(X, W, K)
#if HAVE_TDM
  if (tid < 32) {
    tdm_load_2d((unsigned)(size_t)&sA[0][0], X + (size_t)rowB * K, 64, 128, K);
    tdm_load_2d((unsigned)(size_t)&sB[0][0], W + (size_t)colB * K, 64, 256, K);
  }
#endif
  for (int ks = 0; ks < NK; ++ks) {
#if HAVE_TDM
    if (tid < 32) {
      if (ks + 1 < NK) {
        tdm_load_2d((unsigned)(size_t)&sA[(ks + 1) & 1][0],
                    X + (size_t)rowB * K + (size_t)(ks + 1) * 64, 64, 128, K);
        tdm_load_2d((unsigned)(size_t)&sB[(ks + 1) & 1][0],
                    W + (size_t)colB * K + (size_t)(ks + 1) * 64, 64, 256, K);
        __builtin_amdgcn_s_wait_tensorcnt((short)2);
      } else {
        __builtin_amdgcn_s_wait_tensorcnt((short)0);
      }
    }
    __syncthreads();
#else
    coop_copy_tile(&sA[ks & 1][0], X + (size_t)rowB * K + (size_t)ks * 64, 128, K);
    coop_copy_tile(&sB[ks & 1][0], W + (size_t)colB * K + (size_t)ks * 64, 256, K);
    __syncthreads();
#endif
    const signed char* At = &sA[ks & 1][0];
    const signed char* Bt = &sB[ks & 1][0];
    v8i a[4], b[4];
#pragma unroll
    for (int i = 0; i < 4; ++i) a[i] = load_frag_a(At, 64, rw + 16 * i, 0);
#pragma unroll
    for (int j = 0; j < 4; ++j) b[j] = load_frag_b(Bt, 64, cw + 16 * j, 0);
#pragma unroll
    for (int i = 0; i < 4; ++i)
#pragma unroll
      for (int j = 0; j < 4; ++j) acc[i][j] = wmma_iu8(a[i], b[j], acc[i][j]);
    __syncthreads();
  }
#pragma unroll
  for (int i = 0; i < 4; ++i)
#pragma unroll
    for (int j = 0; j < 4; ++j) {
      int c = colB + cw + 16 * j + m;
      float bterm = (float)bias[c] * LIN_BETA;
#pragma unroll
      for (int e = 0; e < 8; ++e) {
        int r = rowB + rw + 16 * i + e + half8;
        float f = (float)acc[i][j][e] * LIN_ALPHA + bterm;
        out[(size_t)r * N + c] = (signed char)sat8(f);
      }
    }
}

__global__ void __launch_bounds__(256)
gemm_s8_f32_kernel(const signed char* __restrict__ X,
                   const signed char* __restrict__ W,
                   const float* __restrict__ bias, float* __restrict__ out,
                   int K, int N) {
  GEMM_BODY(X, W, K)
#if HAVE_TDM
  if (tid < 32) {
    tdm_load_2d((unsigned)(size_t)&sA[0][0], X + (size_t)rowB * K, 64, 128, K);
    tdm_load_2d((unsigned)(size_t)&sB[0][0], W + (size_t)colB * K, 64, 256, K);
  }
#endif
  for (int ks = 0; ks < NK; ++ks) {
#if HAVE_TDM
    if (tid < 32) {
      if (ks + 1 < NK) {
        tdm_load_2d((unsigned)(size_t)&sA[(ks + 1) & 1][0],
                    X + (size_t)rowB * K + (size_t)(ks + 1) * 64, 64, 128, K);
        tdm_load_2d((unsigned)(size_t)&sB[(ks + 1) & 1][0],
                    W + (size_t)colB * K + (size_t)(ks + 1) * 64, 64, 256, K);
        __builtin_amdgcn_s_wait_tensorcnt((short)2);
      } else {
        __builtin_amdgcn_s_wait_tensorcnt((short)0);
      }
    }
    __syncthreads();
#else
    coop_copy_tile(&sA[ks & 1][0], X + (size_t)rowB * K + (size_t)ks * 64, 128, K);
    coop_copy_tile(&sB[ks & 1][0], W + (size_t)colB * K + (size_t)ks * 64, 256, K);
    __syncthreads();
#endif
    const signed char* At = &sA[ks & 1][0];
    const signed char* Bt = &sB[ks & 1][0];
    v8i a[4], b[4];
#pragma unroll
    for (int i = 0; i < 4; ++i) a[i] = load_frag_a(At, 64, rw + 16 * i, 0);
#pragma unroll
    for (int j = 0; j < 4; ++j) b[j] = load_frag_b(Bt, 64, cw + 16 * j, 0);
#pragma unroll
    for (int i = 0; i < 4; ++i)
#pragma unroll
      for (int j = 0; j < 4; ++j) acc[i][j] = wmma_iu8(a[i], b[j], acc[i][j]);
    __syncthreads();
  }
#pragma unroll
  for (int i = 0; i < 4; ++i)
#pragma unroll
    for (int j = 0; j < 4; ++j) {
      int c = colB + cw + 16 * j + m;
      float bterm = bias[c];
#pragma unroll
      for (int e = 0; e < 8; ++e) {
        int r = rowB + rw + 16 * i + e + half8;
        out[(size_t)r * N + c] = (float)acc[i][j][e] * OUT_ALPHA + bterm;
      }
    }
}

__global__ void rotary_kernel(signed char* __restrict__ x,
                              const float* __restrict__ cosb,
                              const float* __restrict__ sinb, float in_scale) {
  int idx = blockIdx.x * blockDim.x + threadIdx.x;
  if (idx >= Sd * NHd * 64) return;
  int d = idx & 63;
  int nh = (idx >> 6) & (NHd - 1);
  int s = idx >> 10;
  size_t base = ((size_t)s * NHd + nh) * HDd;
  float xa = (float)x[base + d] * in_scale;
  float xb = (float)x[base + d + 64] * in_scale;
  float ca = cosb[s * HDd + d], sa = sinb[s * HDd + d];
  float cb = cosb[s * HDd + d + 64], sb = sinb[s * HDd + d + 64];
  float ya = xa * ca - xb * sa;
  float yb = xb * cb + xa * sb;
  x[base + d]      = (signed char)sat8(ya / ROT_SCALE);
  x[base + d + 64] = (signed char)sat8(yb / ROT_SCALE);
}

__global__ void transpose_v_kernel(const signed char* __restrict__ v,
                                   signed char* __restrict__ vt) {
  __shared__ signed char tile[64][65];
  int h = blockIdx.x;
  int s0 = blockIdx.y * 64;
  int d0 = blockIdx.z * 64;
  for (int t = threadIdx.x; t < 4096; t += 256) {
    int i = t >> 6, j = t & 63;
    tile[i][j] = v[((size_t)(s0 + i) * NHd + h) * HDd + d0 + j];
  }
  __syncthreads();
  for (int t = threadIdx.x; t < 4096; t += 256) {
    int j = t >> 6, i = t & 63;
    vt[((size_t)h * HDd + d0 + j) * Sd + s0 + i] = tile[i][j];
  }
}

__global__ void __launch_bounds__(256)
attn_kernel(const signed char* __restrict__ q8, const signed char* __restrict__ k8,
            const signed char* __restrict__ vt8, const float* __restrict__ mask,
            signed char* __restrict__ ctx8) {
  extern __shared__ float smem[];
  float* sc = smem;
  signed char* pr = (signed char*)smem + (size_t)16 * Sd * 4;

  int h = blockIdx.y;
  int row0 = blockIdx.x * 16;
  int tid = threadIdx.x, wid = tid >> 5, lane = tid & 31;
  int m = lane & 15, half8 = (lane >> 4) << 3;

  const signed char* qh = q8 + (size_t)h * HDd;
  const signed char* kh = k8 + (size_t)h * HDd;
  const int ldqk = NHd * HDd;

  v8i aq0 = load_frag_a(qh, ldqk, row0, 0);
  v8i aq1 = load_frag_a(qh, ldqk, row0, 64);
#pragma unroll 4
  for (int nt = 0; nt < 16; ++nt) {
    int col0 = wid * 256 + nt * 16;
    v8i acc = zero_v8i();
    v8i b0 = load_frag_b(kh, ldqk, col0, 0);
    v8i b1 = load_frag_b(kh, ldqk, col0, 64);
    acc = wmma_iu8(aq0, b0, acc);
    acc = wmma_iu8(aq1, b1, acc);
    int c = col0 + m;
#pragma unroll
    for (int e = 0; e < 8; ++e) {
      int r = e + half8;
      float s = (float)acc[e] * QK_ALPHA + mask[(size_t)(row0 + r) * Sd + c];
      sc[r * Sd + c] = fmaxf(s, -3.402823466e38f);
    }
  }
  __syncthreads();

#pragma unroll
  for (int rr = 0; rr < 2; ++rr) {
    int r = wid * 2 + rr;
    float mx = -3.402823466e38f;
    for (int j = lane; j < Sd; j += 32) mx = fmaxf(mx, sc[r * Sd + j]);
#pragma unroll
    for (int o = 16; o > 0; o >>= 1) mx = fmaxf(mx, __shfl_xor(mx, o, 32));
    float sum = 0.f;
    for (int j = lane; j < Sd; j += 32) sum += __expf(sc[r * Sd + j] - mx);
#pragma unroll
    for (int o = 16; o > 0; o >>= 1) sum += __shfl_xor(sum, o, 32);
    float inv = 127.0f / sum;
    for (int j = lane; j < Sd; j += 32) {
      float p = __expf(sc[r * Sd + j] - mx) * inv;
      pr[r * Sd + j] = (signed char)(int)rintf(p);
    }
  }
  __syncthreads();

  int d0 = wid * 16;
  const signed char* vth = vt8 + (size_t)h * HDd * Sd;
  v8i acc = zero_v8i();
  for (int k0 = 0; k0 < Sd; k0 += 64) {
    v8i ap = load_frag_a(pr, Sd, 0, k0);
    v8i bv = load_frag_b(vth, Sd, d0, k0);
    acc = wmma_iu8(ap, bv, acc);
  }
#pragma unroll
  for (int e = 0; e < 8; ++e) {
    int r = e + half8;
    int c = d0 + m;
    ctx8[((size_t)(row0 + r) * NHd + h) * HDd + c] =
        (signed char)sat8((float)acc[e] * PV_ALPHA);
  }
}

extern "C" void kernel_launch(void* const* d_in, const int* in_sizes, int n_in,
                              void* d_out, int out_size, void* d_ws, size_t ws_size,
                              hipStream_t stream) {
  (void)in_sizes; (void)n_in; (void)out_size; (void)ws_size;

  const int*   hs   = (const int*)d_in[0];
  const float* cosb = (const float*)d_in[1];
  const float* sinb = (const float*)d_in[2];
  const float* mask = (const float*)d_in[3];
  const int*   Wq   = (const int*)d_in[4];
  const int*   bq   = (const int*)d_in[5];
  const int*   Wk   = (const int*)d_in[6];
  const int*   bk   = (const int*)d_in[7];
  const int*   Wv   = (const int*)d_in[8];
  const int*   bv   = (const int*)d_in[9];
  const int*   Wo   = (const int*)d_in[10];
  const float* bo   = (const float*)d_in[11];
  float* out = (float*)d_out;

  const size_t NMAT = (size_t)Sd * Hd;
  signed char* ws = (signed char*)d_ws;
  signed char* x8   = ws; ws += NMAT;
  signed char* wq8  = ws; ws += NMAT;
  signed char* wk8  = ws; ws += NMAT;
  signed char* wv8  = ws; ws += NMAT;
  signed char* wo8  = ws; ws += NMAT;
  signed char* bq8  = ws; ws += 4096;
  signed char* bk8  = ws; ws += 4096;
  signed char* bv8  = ws; ws += 4096;
  signed char* q8   = ws; ws += NMAT;
  signed char* k8   = ws; ws += NMAT;
  signed char* v8b  = ws; ws += NMAT;
  signed char* vt8  = ws; ws += NMAT;
  signed char* ctx8 = ws; ws += NMAT;

  int nBig = (int)NMAT;
  int gBig = (nBig + 255) / 256;
  pack_i8_kernel<<<gBig, 256, 0, stream>>>(hs, x8, nBig);
  pack_i8_kernel<<<gBig, 256, 0, stream>>>(Wq, wq8, nBig);
  pack_i8_kernel<<<gBig, 256, 0, stream>>>(Wk, wk8, nBig);
  pack_i8_kernel<<<gBig, 256, 0, stream>>>(Wv, wv8, nBig);
  pack_i8_kernel<<<gBig, 256, 0, stream>>>(Wo, wo8, nBig);
  pack_i8_kernel<<<(Hd + 255) / 256, 256, 0, stream>>>(bq, bq8, Hd);
  pack_i8_kernel<<<(Hd + 255) / 256, 256, 0, stream>>>(bk, bk8, Hd);
  pack_i8_kernel<<<(Hd + 255) / 256, 256, 0, stream>>>(bv, bv8, Hd);

  dim3 ggrid(Sd / 128, Hd / 256, 1);
  gemm_s8_req_kernel<<<ggrid, 256, 0, stream>>>(x8, wq8, bq8, q8, Hd, Hd);
  gemm_s8_req_kernel<<<ggrid, 256, 0, stream>>>(x8, wk8, bk8, k8, Hd, Hd);
  gemm_s8_req_kernel<<<ggrid, 256, 0, stream>>>(x8, wv8, bv8, v8b, Hd, Hd);

  int nRot = Sd * NHd * 64;
  rotary_kernel<<<(nRot + 255) / 256, 256, 0, stream>>>(q8, cosb, sinb, Q_OUT_SCALE);
  rotary_kernel<<<(nRot + 255) / 256, 256, 0, stream>>>(k8, cosb, sinb, K_OUT_SCALE);

  dim3 tgrid(NHd, Sd / 64, HDd / 64);
  transpose_v_kernel<<<tgrid, 256, 0, stream>>>(v8b, vt8);

  dim3 agrid(Sd / 16, NHd, 1);
  size_t smem = (size_t)16 * Sd * 4 + (size_t)16 * Sd;
  attn_kernel<<<agrid, 256, smem, stream>>>(q8, k8, vt8, mask, ctx8);

  gemm_s8_f32_kernel<<<ggrid, 256, 0, stream>>>(ctx8, wo8, bo, out, Hd, Hd);
}
